// GCN_13821204759316
// MI455X (gfx1250) — matrix operand order repaired and checked by reading, not compiled, not run on hardware
//
#include <hip/hip_runtime.h>
#include <hip/hip_bf16.h>
#include <cstdint>

typedef __bf16 bf16_t;
typedef __attribute__((ext_vector_type(16))) __bf16 v16bf;
typedef __attribute__((ext_vector_type(8)))  __bf16 v8bf;
typedef __attribute__((ext_vector_type(8)))  float  v8f;

#define N_NODES 1024
#define C_CH    256
#define BATCH   8

__device__ __forceinline__ void split_bf16(float v, bf16_t& h, bf16_t& l) {
    h = (bf16_t)v;                 // RNE to bf16
    l = (bf16_t)(v - (float)h);    // residual
}

// A fragment: 16x32 bf16, row-major source with given row stride (bf16 elems).
// lane l (lm=l&15, half=l>>4): row = m0+lm, K = half*16 .. half*16+15 contiguous
__device__ __forceinline__ v16bf load_frag_a(const bf16_t* __restrict__ base,
                                             int row_stride, int m0, int lane) {
    const int lm = lane & 15, half = lane >> 4;
    const bf16_t* r = base + (size_t)(m0 + lm) * row_stride + half * 16;
    v8bf p0 = *(const v8bf*)(r);
    v8bf p1 = *(const v8bf*)(r + 8);
    return __builtin_shufflevector(p0, p1, 0,1,2,3,4,5,6,7,8,9,10,11,12,13,14,15);
}

// B fragment: 32x16 bf16 from an LDS tile stored [N][32] (K contiguous per column).
// lane l: col = n0+lm, K = half*16 .. half*16+15 contiguous.
__device__ __forceinline__ v16bf load_frag_b(const bf16_t* __restrict__ base,
                                             int n0, int lane) {
    const int lm = lane & 15, half = lane >> 4;
    return *(const v16bf*)(base + (size_t)(n0 + lm) * 32 + half * 16);
}

#define WMMA_BF16(A, B, C) \
    __builtin_amdgcn_wmma_f32_16x16x32_bf16(false, (A), false, (B), (short)0, (C), false, false)

// ---------------------------------------------------------------------------
// Kernel 1: d[b,i] = rsqrt(sum_j ind[b,j,i] + 1e-8)   (column sums of ind)
// ---------------------------------------------------------------------------
__global__ __launch_bounds__(256) void degree_kernel(const float* __restrict__ ind,
                                                     float* __restrict__ d) {
    const int b = blockIdx.y;
    const int i = blockIdx.x * 256 + threadIdx.x;
    const float* Ab = ind + (size_t)b * N_NODES * N_NODES;
    float s = 0.f;
    for (int j = 0; j < N_NODES; ++j) s += Ab[(size_t)j * N_NODES + i];
    d[b * N_NODES + i] = rsqrtf(s + 1e-8f);
}

// ---------------------------------------------------------------------------
// Kernel 2: conv weight prep -> bf16 hi/lo in [tap][co][ci] layout
// ---------------------------------------------------------------------------
__global__ __launch_bounds__(256) void wprep_kernel(const float* __restrict__ conv_w,
                                                    bf16_t* __restrict__ w_hi,
                                                    bf16_t* __restrict__ w_lo) {
    const int idx = blockIdx.x * 256 + threadIdx.x;   // tap*65536 + co*256 + ci
    if (idx >= 9 * 256 * 256) return;
    const int ci  = idx & 255;
    const int co  = (idx >> 8) & 255;
    const int tap = idx >> 16;
    const float v = conv_w[(size_t)(co * 256 + ci) * 9 + tap];
    bf16_t h, l; split_bf16(v, h, l);
    w_hi[idx] = h; w_lo[idx] = l;
}

// ---------------------------------------------------------------------------
// Kernel 3: GCN aggregation GEMM (per batch): C[ch,i] = sum_j (x[ch,j]*d[j]) * ind[j,i]
// epilogue: z = d[i]*C + d[i]^2 * x[ch,i]  -> bf16 hi/lo
// Workgroup tile: 128(ch) x 128(i), Ktile = 32, 8 waves arranged 4(M) x 2(N).
// ---------------------------------------------------------------------------
__global__ __launch_bounds__(256) void gcn_agg_kernel(const float* __restrict__ x,
                                                      const float* __restrict__ ind,
                                                      const float* __restrict__ d,
                                                      bf16_t* __restrict__ z_hi,
                                                      bf16_t* __restrict__ z_lo) {
    __shared__ __align__(64) bf16_t sa_hi[128 * 32];
    __shared__ __align__(64) bf16_t sa_lo[128 * 32];
    __shared__ __align__(64) bf16_t sb_hi[128 * 32];   // [i][j] transposed
    __shared__ __align__(64) bf16_t sb_lo[128 * 32];

    const int b       = blockIdx.z;
    const int ch_base = blockIdx.y * 128;
    const int i_base  = blockIdx.x * 128;
    const int tid  = threadIdx.x;
    const int lane = tid & 31, wave = tid >> 5;
    const int mw = wave >> 1;   // 0..3 -> 32 M rows
    const int nw = wave & 1;    // 0..1 -> 64 N cols

    const float* xb = x   + (size_t)b * C_CH * N_NODES;
    const float* Ab = ind + (size_t)b * N_NODES * N_NODES;
    const float* db = d   + (size_t)b * N_NODES;

    v8f acc[2][4] = {};

    for (int k0 = 0; k0 < N_NODES; k0 += 32) {
        __syncthreads();
        {   // stage A: 128 ch x 32 j, scaled by d[j], split hi/lo
            const int c  = tid & 31;       // j offset
            const int r0 = tid >> 5;       // 0..7
            const float dj = db[k0 + c];
            for (int r = r0; r < 128; r += 8) {
                float v = xb[(size_t)(ch_base + r) * N_NODES + k0 + c] * dj;
                bf16_t h, l; split_bf16(v, h, l);
                sa_hi[r * 32 + c] = h;
                sa_lo[r * 32 + c] = l;
            }
        }
        {   // stage B transposed: LDS[i][j] <- ind[j][i]
            const int c  = tid & 127;      // i offset
            const int r0 = tid >> 7;       // 0..1
            for (int r = r0; r < 32; r += 2) {
                float v = Ab[(size_t)(k0 + r) * N_NODES + i_base + c];
                bf16_t h, l; split_bf16(v, h, l);
                sb_hi[c * 32 + r] = h;
                sb_lo[c * 32 + r] = l;
            }
        }
        __syncthreads();

        v16bf ah[2], al[2];
        for (int ms = 0; ms < 2; ++ms) {
            const int m0 = mw * 32 + ms * 16;
            ah[ms] = load_frag_a(sa_hi, 32, m0, lane);
            al[ms] = load_frag_a(sa_lo, 32, m0, lane);
        }
        for (int ns = 0; ns < 4; ++ns) {
            const int n0 = nw * 64 + ns * 16;
            v16bf bh = load_frag_b(sb_hi, n0, lane);
            v16bf bl = load_frag_b(sb_lo, n0, lane);
            for (int ms = 0; ms < 2; ++ms) {
                acc[ms][ns] = WMMA_BF16(ah[ms], bh, acc[ms][ns]);  // hi*hi
                acc[ms][ns] = WMMA_BF16(al[ms], bh, acc[ms][ns]);  // lo*hi
                acc[ms][ns] = WMMA_BF16(ah[ms], bl, acc[ms][ns]);  // hi*lo
            }
        }
    }

    // epilogue: z[ch,i] = d[i]*acc + d[i]^2 * x[ch,i], split into bf16 hi/lo
    const int lm = lane & 15, half = lane >> 4;
    for (int ms = 0; ms < 2; ++ms) {
        for (int ns = 0; ns < 4; ++ns) {
            const int i  = i_base + nw * 64 + ns * 16 + lm;
            const float di = db[i];
            for (int v = 0; v < 8; ++v) {
                const int ch = ch_base + mw * 32 + ms * 16 + v + half * 8;
                const float xv = xb[(size_t)ch * N_NODES + i];
                const float o  = di * acc[ms][ns][v] + di * di * xv;
                bf16_t h, l; split_bf16(o, h, l);
                const size_t zi = ((size_t)b * C_CH + ch) * N_NODES + i;
                z_hi[zi] = h; z_lo[zi] = l;
            }
        }
    }
}

// ---------------------------------------------------------------------------
// Kernel 4: 3x3 conv as 9 shifted GEMMs. One workgroup per (b, row y, co-half).
// Tile: 128(co) x 32(x). LDS holds 3 rows x 34 padded cols x 32 ci (hi/lo).
// ---------------------------------------------------------------------------
__global__ __launch_bounds__(256) void conv3x3_kernel(const bf16_t* __restrict__ z_hi,
                                                      const bf16_t* __restrict__ z_lo,
                                                      const bf16_t* __restrict__ w_hi,
                                                      const bf16_t* __restrict__ w_lo,
                                                      const float* __restrict__ conv_b,
                                                      float* __restrict__ y_out) {
    __shared__ __align__(64) bf16_t szh[3][34][32];   // [dy][x+1][ci]
    __shared__ __align__(64) bf16_t szl[3][34][32];

    const int b       = blockIdx.z;
    const int yrow    = blockIdx.y;
    const int co_base = blockIdx.x * 128;
    const int tid  = threadIdx.x;
    const int lane = tid & 31, wave = tid >> 5;
    const int mw = wave >> 1;   // 0..3 -> 32 co rows
    const int nw = wave & 1;    // 0..1 -> 16 x cols
    const int lm = lane & 15, half = lane >> 4;

    // zero the padding columns (x = -1 and x = 32) once
    if (tid < 192) {
        const int dy    = tid / 64;
        const int which = (tid >> 5) & 1;
        const int ci    = tid & 31;
        const int xp    = which ? 33 : 0;
        szh[dy][xp][ci] = (bf16_t)0.f;
        szl[dy][xp][ci] = (bf16_t)0.f;
    }

    v8f acc[2] = {};
    const size_t zb = (size_t)b * C_CH * N_NODES;

    for (int ci0 = 0; ci0 < 256; ci0 += 32) {
        __syncthreads();
        {   // stage 3 rows of 32 ci x 32 x (zero-fill out-of-range rows)
            const int xcol = tid & 31;
            const int r0   = tid >> 5;  // 0..7
            for (int dy = 0; dy < 3; ++dy) {
                const int row = yrow + dy - 1;
                const bool ok = (row >= 0) && (row < 32);
                for (int r = r0; r < 32; r += 8) {
                    const size_t gi = zb + (size_t)(ci0 + r) * N_NODES + row * 32 + xcol;
                    szh[dy][xcol + 1][r] = ok ? z_hi[gi] : (bf16_t)0.f;
                    szl[dy][xcol + 1][r] = ok ? z_lo[gi] : (bf16_t)0.f;
                }
            }
        }
        __syncthreads();

        for (int dy = 0; dy < 3; ++dy) {
            for (int dx = 0; dx < 3; ++dx) {
                const int tap = dy * 3 + dx;
                const bf16_t* wth = w_hi + (size_t)tap * 256 * 256 + ci0;
                const bf16_t* wtl = w_lo + (size_t)tap * 256 * 256 + ci0;
                // B fragment: column x = nw*16+lm shifted by dx-1 -> padded idx x+dx
                v16bf bh = *(const v16bf*)&szh[dy][nw * 16 + lm + dx][half * 16];
                v16bf bl = *(const v16bf*)&szl[dy][nw * 16 + lm + dx][half * 16];
                for (int ms = 0; ms < 2; ++ms) {
                    const int m0 = co_base + mw * 32 + ms * 16;
                    v16bf ah = load_frag_a(wth, 256, m0, lane);
                    v16bf al = load_frag_a(wtl, 256, m0, lane);
                    acc[ms] = WMMA_BF16(ah, bh, acc[ms]);
                    acc[ms] = WMMA_BF16(al, bh, acc[ms]);
                    acc[ms] = WMMA_BF16(ah, bl, acc[ms]);
                }
            }
        }
    }

    // epilogue: add conv bias, write f32 pre-BN activations
    const int x = nw * 16 + lm;
    for (int ms = 0; ms < 2; ++ms) {
        for (int v = 0; v < 8; ++v) {
            const int co = co_base + mw * 32 + ms * 16 + v + half * 8;
            const float o = acc[ms][v] + conv_b[co];
            y_out[((size_t)b * C_CH + co) * N_NODES + yrow * 32 + x] = o;
        }
    }
}

// ---------------------------------------------------------------------------
// Kernel 5: per-channel batch stats -> scale/shift
// ---------------------------------------------------------------------------
__global__ __launch_bounds__(256) void bnstats_kernel(const float* __restrict__ y,
                                                      const float* __restrict__ gamma,
                                                      const float* __restrict__ beta,
                                                      float* __restrict__ scale,
                                                      float* __restrict__ shift) {
    const int ch = blockIdx.x;
    const int tid = threadIdx.x;
    float s = 0.f, s2 = 0.f;
    for (int b = 0; b < BATCH; ++b) {
        const float* p = y + ((size_t)b * C_CH + ch) * N_NODES;
        for (int i = tid; i < N_NODES; i += 256) {
            const float v = p[i];
            s += v; s2 += v * v;
        }
    }
    __shared__ float rs[256], rs2[256];
    rs[tid] = s; rs2[tid] = s2;
    __syncthreads();
    for (int off = 128; off > 0; off >>= 1) {
        if (tid < off) { rs[tid] += rs[tid + off]; rs2[tid] += rs2[tid + off]; }
        __syncthreads();
    }
    if (tid == 0) {
        const float inv_n = 1.f / (BATCH * N_NODES);
        const float mean = rs[0] * inv_n;
        const float var  = rs2[0] * inv_n - mean * mean;
        const float sc   = gamma[ch] * rsqrtf(var + 1e-5f);
        scale[ch] = sc;
        shift[ch] = beta[ch] - mean * sc;
    }
}

// ---------------------------------------------------------------------------
// Kernel 6: BN apply + ReLU
// ---------------------------------------------------------------------------
__global__ __launch_bounds__(256) void bnapply_kernel(const float* __restrict__ y,
                                                      const float* __restrict__ scale,
                                                      const float* __restrict__ shift,
                                                      float* __restrict__ out) {
    const size_t idx = (size_t)blockIdx.x * 256 + threadIdx.x;
    const int ch = (int)((idx >> 10) & 255);
    const float v = y[idx] * scale[ch] + shift[ch];
    out[idx] = v > 0.f ? v : 0.f;
}

// ---------------------------------------------------------------------------
extern "C" void kernel_launch(void* const* d_in, const int* in_sizes, int n_in,
                              void* d_out, int out_size, void* d_ws, size_t ws_size,
                              hipStream_t stream) {
    const float* x      = (const float*)d_in[0];  // (8,256,32,32)
    const float* ind    = (const float*)d_in[1];  // (8,1024,32,32)
    const float* conv_w = (const float*)d_in[2];  // (256,256,3,3)
    const float* conv_b = (const float*)d_in[3];  // (256,)
    const float* gamma  = (const float*)d_in[4];  // (256,)
    const float* beta   = (const float*)d_in[5];  // (256,)
    float* out = (float*)d_out;                   // (8,256,32,32)

    char* w = (char*)d_ws;
    size_t off = 0;
    auto alloc = [&](size_t bytes) {
        void* p = w + off;
        off = (off + bytes + 255) & ~(size_t)255;
        return p;
    };
    float*  d_deg  = (float*)alloc(BATCH * N_NODES * sizeof(float));
    bf16_t* z_hi   = (bf16_t*)alloc((size_t)BATCH * C_CH * N_NODES * 2);
    bf16_t* z_lo   = (bf16_t*)alloc((size_t)BATCH * C_CH * N_NODES * 2);
    bf16_t* w_hi   = (bf16_t*)alloc((size_t)9 * 256 * 256 * 2);
    bf16_t* w_lo   = (bf16_t*)alloc((size_t)9 * 256 * 256 * 2);
    float*  y_ws   = (float*)alloc((size_t)BATCH * C_CH * N_NODES * sizeof(float));
    float*  scale  = (float*)alloc(256 * sizeof(float));
    float*  shift  = (float*)alloc(256 * sizeof(float));
    (void)ws_size; (void)in_sizes; (void)n_in; (void)out_size;

    degree_kernel<<<dim3(N_NODES / 256, BATCH), 256, 0, stream>>>(ind, d_deg);
    wprep_kernel<<<dim3((9 * 256 * 256 + 255) / 256), 256, 0, stream>>>(conv_w, w_hi, w_lo);
    gcn_agg_kernel<<<dim3(N_NODES / 128, C_CH / 128, BATCH), 256, 0, stream>>>(
        x, ind, d_deg, z_hi, z_lo);
    conv3x3_kernel<<<dim3(C_CH / 128, 32, BATCH), 256, 0, stream>>>(
        z_hi, z_lo, w_hi, w_lo, conv_b, y_ws);
    bnstats_kernel<<<dim3(256), 256, 0, stream>>>(y_ws, gamma, beta, scale, shift);
    bnapply_kernel<<<dim3((BATCH * C_CH * N_NODES) / 256), 256, 0, stream>>>(
        y_ws, scale, shift, out);
}
